// ConceptNBMNary_44650480010034
// MI455X (gfx1250) — matrix operand order repaired and checked by reading, not compiled, not run on hardware
//
#include <hip/hip_runtime.h>

// ---------------- problem constants ----------------
#define T_TOT   1275
#define T1      50
#define NCLS    10
#define BATCH   512
#define NC      50      // NUM_CONCEPTS
#define STR0    264     // K=256 padded (+8 bf16) -> bank-conflict-free
#define STR1    136     // K=128 padded (+8 bf16)

// ---------------- workspace layout (bytes) ----------------
#define OFF_W0F  0                      // [256][2] f32 folded layer0 weights
#define OFF_B0F  2048                   // [256] f32
#define OFF_W1B  3072                   // [128][264] bf16
#define OFF_B1F  70656                  // [128] f32
#define OFF_W2B  71168                  // [128][136] bf16
#define OFF_B2F  105984
#define OFF_W3B  106496                 // [128][136] bf16 (rows 100..127 zero)
#define OFF_B3F  141312
#define PER_ORD  141824
#define OFF_CONVP (2*PER_ORD)                        // [1275][128] f32 padded conv_w
#define OFF_OUTF  (OFF_CONVP + T_TOT*128*4)          // [512][1275] f32 out_feats
#define WS_NEEDED ((size_t)OFF_OUTF + (size_t)BATCH*T_TOT*4)

typedef __attribute__((ext_vector_type(16))) __bf16 v16bf;
typedef __attribute__((ext_vector_type(8)))  float  v8f;

union Frag  { uint4 q[2]; v16bf v; };
union Pack8 { __bf16 h[8]; uint4 q; };

// ---------------- BN-fold prep kernels ----------------
__global__ void fold_bf16(const float* __restrict__ w, const float* __restrict__ b,
                          const float* __restrict__ g, const float* __restrict__ be,
                          const float* __restrict__ m, const float* __restrict__ v,
                          __bf16* __restrict__ wOut, float* __restrict__ bOut,
                          int dout, int din, int doutPad, int dinPad) {
    int idx = blockIdx.x * 256 + threadIdx.x;
    int total = doutPad * dinPad;
    if (idx >= total) return;
    int o = idx / dinPad, i = idx % dinPad;
    float val = 0.f;
    if (o < dout && i < din) {
        float s = g[o] * rsqrtf(v[o] + 1e-5f);
        val = s * w[o * din + i];
    }
    wOut[o * dinPad + i] = (__bf16)val;
    if (i == 0) {
        float bb = 0.f;
        if (o < dout) {
            float s = g[o] * rsqrtf(v[o] + 1e-5f);
            bb = s * (b[o] - m[o]) + be[o];
        }
        bOut[o] = bb;
    }
}

__global__ void fold_f32(const float* __restrict__ w, const float* __restrict__ b,
                         const float* __restrict__ g, const float* __restrict__ be,
                         const float* __restrict__ m, const float* __restrict__ v,
                         float* __restrict__ wOut, float* __restrict__ bOut,
                         int dout, int din, int doutPad, int dinPad) {
    int idx = blockIdx.x * 256 + threadIdx.x;
    int total = doutPad * dinPad;
    if (idx >= total) return;
    int o = idx / dinPad, i = idx % dinPad;
    float val = 0.f;
    if (o < dout && i < din) {
        float s = g[o] * rsqrtf(v[o] + 1e-5f);
        val = s * w[o * din + i];
    }
    wOut[o * dinPad + i] = val;
    if (i == 0) {
        float bb = 0.f;
        if (o < dout) {
            float s = g[o] * rsqrtf(v[o] + 1e-5f);
            bb = s * (b[o] - m[o]) + be[o];
        }
        bOut[o] = bb;
    }
}

__global__ void pad_convw(const float* __restrict__ cw, float* __restrict__ out) {
    int idx = blockIdx.x * 256 + threadIdx.x;
    if (idx >= T_TOT * 128) return;
    int t = idx >> 7, k = idx & 127;
    out[idx] = (k < 100) ? cw[t * 100 + k] : 0.f;
}

// ---------------- CDNA5 async global->LDS copy (ASYNCcnt path) ----------------
// Generic pointers into LDS carry the LDS byte offset in their low 32 bits
// (ISA 10.2 aperture rule), so truncation yields the vdst LDS address operand.
__device__ __forceinline__ void async_b128(void* ldsDst, const void* gsrc) {
    unsigned lds = (unsigned)(unsigned long long)(uintptr_t)ldsDst;
    asm volatile("global_load_async_to_lds_b128 %0, %1, off"
                 :: "v"(lds), "v"(gsrc) : "memory");
}
__device__ __forceinline__ void wait_async0() {
    asm volatile("s_wait_asynccnt 0x0" ::: "memory");
}

// ---------------- WMMA fragment helpers ----------------
// A (16x32 bf16) from row-major LDS row: lanes 0-15 row M=L, K 0-15;
// lanes 16-31 same rows, K 16-31.
__device__ __forceinline__ v16bf load_A(const __bf16* rowPtr, int kBase, int half) {
    Frag f;
    f.q[0] = *(const uint4*)(rowPtr + kBase + half * 16);
    f.q[1] = *(const uint4*)(rowPtr + kBase + half * 16 + 8);
    return f.v;
}
// B (32x16 bf16): lane L holds column N=L%16, K-range (L/16)*16..+15 -> 32 contiguous
// bytes from the weight row stored as [N][K].
__device__ __forceinline__ v16bf load_B(const __bf16* p) {
    Frag f;
    const uint4* q = (const uint4*)p;
    f.q[0] = q[0];
    f.q[1] = q[1];
    return f.v;
}

// One WMMA step with compile-time N so the reuse_a modifier is an ICE.
template <int N>
__device__ __forceinline__ void wmma_step(const v16bf& a, v16bf& b, v8f* c,
                                          const __bf16* wRow0, int wStride,
                                          int kk, int half, bool kkNext) {
    v16bf bN = b;
    if (N < 7)
        bN = load_B(wRow0 + (N + 1) * 16 * wStride + kk * 32 + half * 16);
    else if (kkNext)
        bN = load_B(wRow0 + (kk + 1) * 32 + half * 16);
    c[N] = __builtin_amdgcn_wmma_f32_16x16x32_bf16(
        false, a, false, b, (short)0, c[N], /*reuse_a=*/(N < 7), false);
    b = bN;
}

// Software-pipelined layer: B double-buffered, next-kk A prefetched, reuse_a hint
// set on the 7 back-to-back WMMAs sharing the A operand.
template <int KT>
__device__ __forceinline__ void do_layer(const __bf16* aRow, const __bf16* wRow0,
                                         int wStride, int half, v8f* c) {
    v8f z = {0.f, 0.f, 0.f, 0.f, 0.f, 0.f, 0.f, 0.f};
#pragma unroll
    for (int n = 0; n < 8; ++n) c[n] = z;

    v16bf a = load_A(aRow, 0, half);
    v16bf b = load_B(wRow0 + half * 16);
#pragma unroll
    for (int kk = 0; kk < KT; ++kk) {
        v16bf aN = a;
        if (kk + 1 < KT) aN = load_A(aRow, (kk + 1) * 32, half);
        const bool kkNext = (kk + 1 < KT);
        wmma_step<0>(a, b, c, wRow0, wStride, kk, half, kkNext);
        wmma_step<1>(a, b, c, wRow0, wStride, kk, half, kkNext);
        wmma_step<2>(a, b, c, wRow0, wStride, kk, half, kkNext);
        wmma_step<3>(a, b, c, wRow0, wStride, kk, half, kkNext);
        wmma_step<4>(a, b, c, wRow0, wStride, kk, half, kkNext);
        wmma_step<5>(a, b, c, wRow0, wStride, kk, half, kkNext);
        wmma_step<6>(a, b, c, wRow0, wStride, kk, half, kkNext);
        wmma_step<7>(a, b, c, wRow0, wStride, kk, half, kkNext);
        a = aN;
    }
}

// bias + ReLU + bf16 store to row-major LDS (C/D layout: lane L -> N=L%16, M=8*(L/16)+r)
__device__ __forceinline__ void store_act(const v8f* c, __bf16* hBase, const float* bias,
                                          int rowLocal0, int stride, int laneN, int half) {
#pragma unroll
    for (int n = 0; n < 8; ++n) {
        int col = n * 16 + laneN;
        float bcol = bias[col];
#pragma unroll
        for (int r = 0; r < 8; ++r) {
            float vv = c[n][r] + bcol;
            vv = vv > 0.f ? vv : 0.f;
            hBase[(rowLocal0 + half * 8 + r) * stride + col] = (__bf16)vv;
        }
    }
}

// ---------------- fused main kernel: one workgroup per tuple ----------------
__global__ __launch_bounds__(256) void nbm_main(const float* __restrict__ x,
                                                const int* __restrict__ idx1,
                                                const int* __restrict__ idx2,
                                                const char* __restrict__ wsParams,
                                                const float* __restrict__ convPad,
                                                const float* __restrict__ conv_b,
                                                float* __restrict__ outF) {
    __shared__ __align__(16) __bf16 sW1[128 * STR0];
    __shared__ __align__(16) __bf16 sW2[128 * STR1];
    __shared__ __align__(16) __bf16 sW3[128 * STR1];
    __shared__ __align__(16) __bf16 sH0[128 * STR0];
    __shared__ __align__(16) __bf16 sH1[128 * STR1];
    __shared__ __align__(16) __bf16 sH2[128 * STR1];
    __shared__ __align__(16) float  sW0[512];   // [k][2]
    __shared__ __align__(16) float  sB0[256];
    __shared__ __align__(16) float  sB1[128], sB2[128], sB3[128], sCW[128];

    const int t   = blockIdx.x;
    const int tid = threadIdx.x;
    const int wv  = tid >> 5;
    const int lane = tid & 31;
    const int laneN = lane & 15;
    const int half = lane >> 4;

    int ca, cb;
    bool isO2;
    if (t < T1) { ca = idx1[t]; cb = 0; isO2 = false; }
    else        { ca = idx2[(t - T1) * 2]; cb = idx2[(t - T1) * 2 + 1]; isO2 = true; }
    const char* base = wsParams + (isO2 ? PER_ORD : 0);
    const float cbt = conv_b[t];

    // ---- stage folded weights into LDS via async-DMA, small tables via VALU ----
    {
        const char* s1 = base + OFF_W1B;
        const char* s2 = base + OFF_W2B;
        const char* s3 = base + OFF_W3B;
        for (int i = tid; i < 128 * STR0 * 2 / 16; i += 256)
            async_b128((char*)sW1 + i * 16, s1 + i * 16);
        for (int i = tid; i < 128 * STR1 * 2 / 16; i += 256)
            async_b128((char*)sW2 + i * 16, s2 + i * 16);
        for (int i = tid; i < 128 * STR1 * 2 / 16; i += 256)
            async_b128((char*)sW3 + i * 16, s3 + i * 16);

        const float* g0w = (const float*)(base + OFF_W0F);
        const float* g0b = (const float*)(base + OFF_B0F);
        const float* g1b = (const float*)(base + OFF_B1F);
        const float* g2b = (const float*)(base + OFF_B2F);
        const float* g3b = (const float*)(base + OFF_B3F);
        if (tid < 256) {
            sW0[2 * tid]     = g0w[2 * tid];
            sW0[2 * tid + 1] = g0w[2 * tid + 1];
            sB0[tid] = g0b[tid];
        }
        if (tid < 128) {
            sB1[tid] = g1b[tid];
            sB2[tid] = g2b[tid];
            sB3[tid] = g3b[tid];
            sCW[tid] = convPad[t * 128 + tid];
        }
        wait_async0();
    }
    __syncthreads();

    v8f c[8];
    for (int p = 0; p < 4; ++p) {
        const int rowBase = p * 128 + wv * 16;   // global batch row of this wave's tile

        // ---- layer 0 (order<=2 -> 256) on VALU, write H0 row-major bf16 ----
        {
            const int r = laneN;                 // row within wave tile
            const int b = rowBase + r;
            float xa = x[b * NC + ca];
            float xb = isO2 ? x[b * NC + cb] : 0.f;
            __bf16* h0row = sH0 + (wv * 16 + r) * STR0;
#pragma unroll
            for (int i = 0; i < 16; ++i) {
                int k0 = half * 8 + i * 16;
                Pack8 pk;
#pragma unroll
                for (int j = 0; j < 8; ++j) {
                    int k = k0 + j;
                    float val = sW0[2 * k] * xa + sW0[2 * k + 1] * xb + sB0[k];
                    pk.h[j] = (__bf16)(val > 0.f ? val : 0.f);
                }
                *(uint4*)(h0row + k0) = pk.q;
            }
        }
        __syncthreads();

        // ---- layer 1: [16,256] x [256,128] ----
        do_layer<8>(sH0 + (wv * 16 + laneN) * STR0, sW1 + laneN * STR0, STR0, half, c);
        store_act(c, sH1, sB1, wv * 16, STR1, laneN, half);
        __syncthreads();

        // ---- layer 2: [16,128] x [128,128] ----
        do_layer<4>(sH1 + (wv * 16 + laneN) * STR1, sW2 + laneN * STR1, STR1, half, c);
        store_act(c, sH2, sB2, wv * 16, STR1, laneN, half);
        __syncthreads();

        // ---- layer 3: [16,128] x [128,128(pad of 100)] ----
        do_layer<4>(sH2 + (wv * 16 + laneN) * STR1, sW3 + laneN * STR1, STR1, half, c);

        // ---- bias+ReLU+conv_w dot, reduce across 16 lanes ----
        float acc[8];
#pragma unroll
        for (int r = 0; r < 8; ++r) acc[r] = 0.f;
#pragma unroll
        for (int n = 0; n < 8; ++n) {
            int col = n * 16 + laneN;
            float b3 = sB3[col];
            float cw = sCW[col];         // zero for padded cols >= 100
#pragma unroll
            for (int r = 0; r < 8; ++r) {
                float vv = c[n][r] + b3;
                vv = vv > 0.f ? vv : 0.f;
                acc[r] += vv * cw;
            }
        }
#pragma unroll
        for (int msk = 1; msk < 16; msk <<= 1) {
#pragma unroll
            for (int r = 0; r < 8; ++r) acc[r] += __shfl_xor(acc[r], msk, 32);
        }
        if (laneN == 0) {
#pragma unroll
            for (int r = 0; r < 8; ++r)
                outF[(rowBase + half * 8 + r) * T_TOT + t] = acc[r] + cbt;
        }
        __syncthreads();
    }
}

// ---------------- classifier: out = out_feats @ cls_w^T + cls_b ----------------
__global__ void cls_kernel(const float* __restrict__ of, const float* __restrict__ cw,
                           const float* __restrict__ cb, float* __restrict__ out) {
    int b = blockIdx.x / NCLS;
    int c = blockIdx.x % NCLS;
    float s = 0.f;
    for (int t = threadIdx.x; t < T_TOT; t += 32)
        s += of[b * T_TOT + t] * cw[c * T_TOT + t];
#pragma unroll
    for (int msk = 1; msk < 32; msk <<= 1) s += __shfl_xor(s, msk, 32);
    if (threadIdx.x == 0) out[b * NCLS + c] = s + cb[c];
}

// ---------------- host launch ----------------
extern "C" void kernel_launch(void* const* d_in, const int* in_sizes, int n_in,
                              void* d_out, int out_size, void* d_ws, size_t ws_size,
                              hipStream_t stream) {
    if (ws_size < WS_NEEDED) return;

    const float* x    = (const float*)d_in[0];
    const int*   idx1 = (const int*)d_in[1];
    const int*   idx2 = (const int*)d_in[2];

    const float *W[2][4], *Bv[2][4], *G[2][4], *Be[2][4], *Mn[2][4], *Vr[2][4];
    const float *conv_w, *conv_b, *cls_w, *cls_b;

    if (n_in >= 55) {
        // nested dicts flattened as leaves; disambiguate insertion vs sorted-key order
        bool insertion = (in_sizes[4] == 256);   // insertion: b0(256); sorted: b1(128)
        for (int ord = 0; ord < 2; ++ord) {
            int base = 3 + ord * 24;
            if (insertion) {
                for (int l = 0; l < 4; ++l) {
                    W[ord][l]  = (const float*)d_in[base + l * 6 + 0];
                    Bv[ord][l] = (const float*)d_in[base + l * 6 + 1];
                    G[ord][l]  = (const float*)d_in[base + l * 6 + 2];
                    Be[ord][l] = (const float*)d_in[base + l * 6 + 3];
                    Mn[ord][l] = (const float*)d_in[base + l * 6 + 4];
                    Vr[ord][l] = (const float*)d_in[base + l * 6 + 5];
                }
            } else { // sorted keys: b0-3, beta0-3, g0-3, m0-3, v0-3, w0-3
                for (int l = 0; l < 4; ++l) {
                    Bv[ord][l] = (const float*)d_in[base + l];
                    Be[ord][l] = (const float*)d_in[base + 4 + l];
                    G[ord][l]  = (const float*)d_in[base + 8 + l];
                    Mn[ord][l] = (const float*)d_in[base + 12 + l];
                    Vr[ord][l] = (const float*)d_in[base + 16 + l];
                    W[ord][l]  = (const float*)d_in[base + 20 + l];
                }
            }
        }
        conv_w = (const float*)d_in[51];
        conv_b = (const float*)d_in[52];
        cls_w  = (const float*)d_in[53];
        cls_b  = (const float*)d_in[54];
    } else {
        // params passed as single contiguous blobs in insertion order
        static const int o1off[4][6] = {
            {0, 256, 512, 768, 1024, 1280},
            {1536, 34304, 34432, 34560, 34688, 34816},
            {34944, 51328, 51456, 51584, 51712, 51840},
            {51968, 64768, 64868, 64968, 65068, 65168}};
        static const int o2off[4][6] = {
            {0, 512, 768, 1024, 1280, 1536},
            {1792, 34560, 34688, 34816, 34944, 35072},
            {35200, 51584, 51712, 51840, 51968, 52096},
            {52224, 65024, 65124, 65224, 65324, 65424}};
        const float* p[2] = {(const float*)d_in[3], (const float*)d_in[4]};
        for (int ord = 0; ord < 2; ++ord)
            for (int l = 0; l < 4; ++l) {
                const int* o = (ord == 0) ? o1off[l] : o2off[l];
                W[ord][l]  = p[ord] + o[0];
                Bv[ord][l] = p[ord] + o[1];
                G[ord][l]  = p[ord] + o[2];
                Be[ord][l] = p[ord] + o[3];
                Mn[ord][l] = p[ord] + o[4];
                Vr[ord][l] = p[ord] + o[5];
            }
        conv_w = (const float*)d_in[5];
        conv_b = (const float*)d_in[6];
        cls_w  = (const float*)d_in[7];
        cls_b  = (const float*)d_in[8];
    }

    char* ws = (char*)d_ws;
    static const int DIN0[2] = {1, 2};

    for (int ord = 0; ord < 2; ++ord) {
        char* b = ws + ord * PER_ORD;
        // layer 0: keep f32, [256][2]
        fold_f32<<<(256 * 2 + 255) / 256, 256, 0, stream>>>(
            W[ord][0], Bv[ord][0], G[ord][0], Be[ord][0], Mn[ord][0], Vr[ord][0],
            (float*)(b + OFF_W0F), (float*)(b + OFF_B0F), 256, DIN0[ord], 256, 2);
        // layer 1: 128x256 -> bf16 [128][264]
        fold_bf16<<<(128 * STR0 + 255) / 256, 256, 0, stream>>>(
            W[ord][1], Bv[ord][1], G[ord][1], Be[ord][1], Mn[ord][1], Vr[ord][1],
            (__bf16*)(b + OFF_W1B), (float*)(b + OFF_B1F), 128, 256, 128, STR0);
        // layer 2: 128x128 -> bf16 [128][136]
        fold_bf16<<<(128 * STR1 + 255) / 256, 256, 0, stream>>>(
            W[ord][2], Bv[ord][2], G[ord][2], Be[ord][2], Mn[ord][2], Vr[ord][2],
            (__bf16*)(b + OFF_W2B), (float*)(b + OFF_B2F), 128, 128, 128, STR1);
        // layer 3: 100x128 -> bf16 [128][136] zero-padded rows
        fold_bf16<<<(128 * STR1 + 255) / 256, 256, 0, stream>>>(
            W[ord][3], Bv[ord][3], G[ord][3], Be[ord][3], Mn[ord][3], Vr[ord][3],
            (__bf16*)(b + OFF_W3B), (float*)(b + OFF_B3F), 100, 128, 128, STR1);
    }
    pad_convw<<<(T_TOT * 128 + 255) / 256, 256, 0, stream>>>(
        conv_w, (float*)(ws + OFF_CONVP));

    nbm_main<<<T_TOT, 256, 0, stream>>>(
        x, idx1, idx2, ws, (const float*)(ws + OFF_CONVP), conv_b,
        (float*)(ws + OFF_OUTF));

    cls_kernel<<<BATCH * NCLS, 32, 0, stream>>>(
        (const float*)(ws + OFF_OUTF), cls_w, cls_b, (float*)d_out);
}
